// ProLiFRenderer_46703474377271
// MI455X (gfx1250) — matrix operand order repaired and checked by reading, not compiled, not run on hardware
//
#include <hip/hip_runtime.h>

typedef __attribute__((ext_vector_type(16))) _Float16 v16h;
typedef __attribute__((ext_vector_type(2)))  __fp16   v2hf;
typedef __attribute__((ext_vector_type(8)))  float    v8f;

#define HD 128
// Fragments: 8 (W0, bias in row k=3) + 3*40 (hidden, 5 k-steps w/ bias row) + 5 (Wout w/ bias row)
#define NFRAG 133
#define FRAG_HALVES (NFRAG * 512)       // 32 lanes * 16 halves per fragment
#define FRAG_BYTES  (FRAG_HALVES * 2)   // 136192
#define LDS_BYTES   FRAG_BYTES

#define INV2PI 0.15915494309189535f

union V16 {
  v16h     v;
  _Float16 h[16];
  unsigned u[8];
};

__device__ __forceinline__ unsigned shfx16(unsigned x) {
  return (unsigned)__shfl_xor((int)x, 16, 32);
}
__device__ __forceinline__ float fsigmoid(float x) { return 1.0f / (1.0f + __expf(-x)); }

// pack two f32 -> one u32 of two f16 (v_cvt_pk_rtz_f16_f32)
__device__ __forceinline__ unsigned pk16(float a, float b) {
  union { v2hf h; unsigned u; } c;
  c.h = __builtin_amdgcn_cvt_pkrtz(a, b);
  return c.u;
}

// ---------------------------------------------------------------------------
// Kernel 1: swizzle weights (with biases folded in as augmented-K rows) into
// WMMA A-fragment layout (f16) in d_ws.
// A-fragment (16x32 f16): lane<16 holds row M=lane&15, K = e;
// lane>=16 same row, K offset +16.  We store W^T so A rows = out features.
// Fragment table:
//   [0..7]      W0^T  m-tile mt, single k-step; row k=3 holds b0
//   [8..127]    hidden layer l: idx = 8 + l*40 + mt*5 + kk, kk=0..4;
//               kk==4 fragment holds bs[l] in row k=128 (koff==0), rest 0
//   [128..132]  Wout^T: idx = 128 + kk, kk=0..4; kk==4 holds bout in row k=128
// ---------------------------------------------------------------------------
__global__ void prolif_prep(const float* __restrict__ W0, const float* __restrict__ b0,
                            const float* __restrict__ Ws, const float* __restrict__ bs,
                            const float* __restrict__ Wout, const float* __restrict__ bout,
                            _Float16* __restrict__ fragOut) {
  int i = blockIdx.x * blockDim.x + threadIdx.x;
  if (i >= FRAG_HALVES) return;
  int frag = i >> 9;
  int r    = i & 511;
  int lane = r >> 4;
  int e    = r & 15;
  int nloc = lane & 15;
  int koff = e + ((lane >= 16) ? 16 : 0);
  float val = 0.0f;
  if (frag < 8) {                         // W0^T + b0 row at k=3
    int n = frag * 16 + nloc;
    if (koff < 3)       val = W0[koff * HD + n];
    else if (koff == 3) val = b0[n];
  } else if (frag < 128) {                // hidden layer l
    int f  = frag - 8;
    int l  = f / 40;
    int rm = f % 40;
    int mt = rm / 5, kk = rm % 5;
    int n  = mt * 16 + nloc;
    if (kk < 4)          val = Ws[l * HD * HD + (kk * 32 + koff) * HD + n];
    else if (koff == 0)  val = bs[l * HD + n];   // bias row k=128
  } else {                                // Wout^T + bout row
    int kk = frag - 128;
    if (kk < 4) {
      if (nloc < 4) val = Wout[(kk * 32 + koff) * 4 + nloc];
    } else if (koff == 0 && nloc < 4) {
      val = bout[nloc];
    }
  }
  fragOut[i] = (_Float16)val;
}

// ---------------------------------------------------------------------------
// Activation: h = sin(scale*acc) as v_sin(acc*cs), cs = scale/2pi (hardware
// sine takes revolutions; bias already inside acc via the augmented-K WMMA).
// Pack f32->f16 pairs (cvt_pk_rtz) and repack D-layout (M=feature, N=sample)
// into next-layer B fragments (lane<16: K=0..15, lane>=16: K=16..31) via a
// single lane<->lane+16 exchange on packed u32 pairs.
// ---------------------------------------------------------------------------
__device__ __forceinline__ void activate_pack(const v8f (&acc)[8], float cs,
                                              V16 (&hf)[4], bool lo) {
  unsigned actu[8][4];
#pragma unroll
  for (int mt = 0; mt < 8; ++mt) {
#pragma unroll
    for (int i = 0; i < 4; ++i) {
      float s0 = __builtin_amdgcn_sinf(acc[mt][2 * i]     * cs);
      float s1 = __builtin_amdgcn_sinf(acc[mt][2 * i + 1] * cs);
      actu[mt][i] = pk16(s0, s1);
    }
  }
#pragma unroll
  for (int kk = 0; kk < 4; ++kk) {
#pragma unroll
    for (int i = 0; i < 4; ++i) {
      unsigned a0 = actu[2 * kk][i], a1 = actu[2 * kk + 1][i];
      unsigned s0 = shfx16(a0), s1 = shfx16(a1);
      hf[kk].u[i]     = lo ? a0 : s1;
      hf[kk].u[4 + i] = lo ? s0 : a1;
    }
  }
}

// ---------------------------------------------------------------------------
// Kernel 2: fused MLP + volume rendering. 8 waves/block, 1 ray per wave,
// 4 sample-tiles of 16 per ray. Weight fragments (biases included) in LDS;
// all WMMA chains start from inline C = 0.
// ---------------------------------------------------------------------------
__global__ __launch_bounds__(256)
__attribute__((amdgpu_waves_per_eu(4)))
void prolif_render(const float* __restrict__ coords,
                   const float* __restrict__ steps,
                   const float* __restrict__ bg,
                   const unsigned* __restrict__ wsblob,
                   float* __restrict__ out) {
  extern __shared__ __align__(32) char smem[];
  {
    uint4* dst = (uint4*)smem;
    const uint4* src = (const uint4*)wsblob;
    for (int i = threadIdx.x; i < LDS_BYTES / 16; i += 256) dst[i] = src[i];
  }
  __syncthreads();
  const v16h* fragL = (const v16h*)smem;                // [frag*32 + lane]

  const int  lane = threadIdx.x & 31;
  const int  wave = threadIdx.x >> 5;
  const int  ray  = blockIdx.x * 8 + wave;
  const bool lo   = (lane < 16);
  const int  sl   = lane & 15;

  const float ux = coords[ray * 4 + 0], uy = coords[ray * 4 + 1];
  const float sx = coords[ray * 4 + 2], sy = coords[ray * 4 + 3];

  // prefetch all 64 step values for this ray (2 coalesced loads, no per-tile stall)
  const float st0 = steps[ray * 64 + lane];
  const float st1 = steps[ray * 64 + 32 + lane];

  float cr = 0.0f, cg = 0.0f, cb = 0.0f;
  float Tsum = 0.0f;
  const v8f zero8 = {};

  // constant B fragment selecting the bias row: 1.0 at K=128 (kk=4, lane<16 e0)
  V16 ones;
#pragma unroll
  for (int i = 0; i < 8; ++i) ones.u[i] = 0u;
  ones.u[0] = lo ? 0x00003C00u : 0u;

#pragma unroll 1
  for (int tt = 0; tt < 4; ++tt) {
    // ---- build x^T B-fragment: lane<16 sample sl, features [px,py,t,1,0..]
    float tsrc = (tt < 2) ? st0 : st1;
    float t    = __shfl(tsrc, (tt & 1) * 16 + sl, 32);
    t          = lo ? t : 0.0f;
    float px = ux * (1.0f - t) + sx * t;
    float py = uy * (1.0f - t) + sy * t;
    V16 bf;
#pragma unroll
    for (int i = 0; i < 8; ++i) bf.u[i] = 0u;
    bf.u[0] = lo ? pk16(px, py) : 0u;
    bf.u[1] = lo ? pk16(t, 1.0f) : 0u;    // feature 3 == 1.0 -> picks up b0 row

    v8f acc[8];
    // ---- layer 0: D = [W0;b0]^T [x;1]^T  (C = 0)
#pragma unroll
    for (int mt = 0; mt < 8; ++mt) {
      V16 af; af.v = fragL[mt * 32 + lane];
      acc[mt] = __builtin_amdgcn_wmma_f32_16x16x32_f16(false, af.v, false, bf.v,
                                                       (short)0, zero8, false, false);
    }
    V16 hf[4];
    activate_pack(acc, 30.0f * INV2PI, hf, lo);   // h = sin(30*(xW0+b0))

    // ---- 3 hidden layers: D = [Ws;bs]^T [h;1]^T  (5 k-steps, C = 0)
#pragma unroll 1
    for (int l = 0; l < 3; ++l) {
#pragma unroll
      for (int kk = 0; kk < 5; ++kk) {
#pragma unroll
        for (int mt = 0; mt < 8; ++mt) {
          V16 af; af.v = fragL[(8 + l * 40 + mt * 5 + kk) * 32 + lane];
          acc[mt] = __builtin_amdgcn_wmma_f32_16x16x32_f16(
              false, af.v, false, (kk == 4) ? ones.v : hf[kk].v, (short)0,
              (kk == 0) ? zero8 : acc[mt], false, false);
        }
      }
      activate_pack(acc, INV2PI, hf, lo);
    }

    // ---- output layer: rgba^T = [Wout;bout]^T [h;1]^T (5 k-steps, C = 0)
    v8f accO = zero8;
#pragma unroll
    for (int kk = 0; kk < 5; ++kk) {
      V16 af; af.v = fragL[(128 + kk) * 32 + lane];
      accO = __builtin_amdgcn_wmma_f32_16x16x32_f16(
          false, af.v, false, (kk == 4) ? ones.v : hf[kk].v, (short)0,
          (kk == 0) ? zero8 : accO, false, false);
    }

    // ---- volume rendering for this tile (lane<16 holds sample sl's rgba)
    float r = fsigmoid(accO[0]);
    float g = fsigmoid(accO[1]);
    float b = fsigmoid(accO[2]);
    float z = 10.0f * accO[3];                               // BETA * a
    float sp = (z > 20.0f) ? z : __logf(1.0f + __expf(z));   // softplus
    float delta = lo ? (sp * 0.1f) : 0.0f;                   // density * SCALE

    float incl = delta;                                      // in-order prefix over 16 samples
#pragma unroll
    for (int d = 1; d < 16; d <<= 1) {
      float vsh = __shfl_up(incl, d, 32);
      if (lane >= d) incl += vsh;
    }
    float excl = incl - delta;
    float w = (1.0f - __expf(-delta)) * __expf(-(Tsum + excl));
    w = lo ? w : 0.0f;
    cr += r * w; cg += g * w; cb += b * w;
    Tsum += __shfl(incl, 15, 32);                            // carry optical depth
  }

  float Tfin = __expf(-Tsum);
#pragma unroll
  for (int off = 8; off; off >>= 1) {
    cr += __shfl_down(cr, off, 32);
    cg += __shfl_down(cg, off, 32);
    cb += __shfl_down(cb, off, 32);
  }
  if (lane == 0) {
    out[ray * 3 + 0] = cr + bg[ray * 3 + 0] * Tfin;
    out[ray * 3 + 1] = cg + bg[ray * 3 + 1] * Tfin;
    out[ray * 3 + 2] = cb + bg[ray * 3 + 2] * Tfin;
  }
}

extern "C" void kernel_launch(void* const* d_in, const int* in_sizes, int n_in,
                              void* d_out, int out_size, void* d_ws, size_t ws_size,
                              hipStream_t stream) {
  const float* coords = (const float*)d_in[0];
  const float* steps  = (const float*)d_in[1];
  const float* bg     = (const float*)d_in[2];
  const float* W0     = (const float*)d_in[3];
  const float* b0     = (const float*)d_in[4];
  const float* Ws     = (const float*)d_in[5];
  const float* bs     = (const float*)d_in[6];
  const float* Wout   = (const float*)d_in[7];
  const float* bout   = (const float*)d_in[8];
  float* out = (float*)d_out;

  _Float16* fragOut = (_Float16*)d_ws;

  prolif_prep<<<(FRAG_HALVES + 255) / 256, 256, 0, stream>>>(W0, b0, Ws, bs, Wout, bout,
                                                             fragOut);

  (void)hipFuncSetAttribute(reinterpret_cast<const void*>(prolif_render),
                            hipFuncAttributeMaxDynamicSharedMemorySize, LDS_BYTES);
  prolif_render<<<8192 / 8, 256, LDS_BYTES, stream>>>(coords, steps, bg,
                                                      (const unsigned*)d_ws, out);
}
